// MLP_41506563948564
// MI455X (gfx1250) — compile-verified, matrix operand order checked by reading, not run on hardware
//
#include <hip/hip_runtime.h>
#include <math.h>
#include <stdint.h>

// Problem constants (from the reference)
#define Bb 2
#define Ss 2048
#define Dd 1024
#define Hh 2048
#define Ee 8
#define Kk 2
#define Tt (Bb * Ss)              // 4096 tokens
#define Pp (Tt * Kk)              // 8192 (token, expert) pairs

#define MT 32                     // M rows per block (2 WMMA tiles)
#define NB 128                    // N columns per block (8 waves x 16)
#define KC 32                     // K per WMMA step
#define RS 40                     // LDS row stride in halves (80B: 16B-aligned, bank-spread)
#define MAX_TILES (Pp / MT + Ee)  // 264 worst-case 32-row tiles

typedef unsigned short u16;
typedef __attribute__((ext_vector_type(16))) __bf16 bf16x16;
typedef __attribute__((ext_vector_type(8)))  float  floatx8;

union Frag { bf16x16 v; uint4 q[2]; };

__device__ __forceinline__ u16 f2bf(float f) {
  union { float f; unsigned u; } c; c.f = f;
  unsigned u = c.u;
  unsigned r = u + 0x7FFFu + ((u >> 16) & 1u);   // round-to-nearest-even
  return (u16)(r >> 16);
}

// CDNA5 async global->LDS copies (ASYNCcnt-tracked)
__device__ __forceinline__ void async_b128(unsigned lds, const void* gp) {
  asm volatile("global_load_async_to_lds_b128 %0, %1, off"
               :: "v"(lds), "v"((unsigned long long)(uintptr_t)gp) : "memory");
}
__device__ __forceinline__ void async_b64(unsigned lds, const void* gp) {
  asm volatile("global_load_async_to_lds_b64 %0, %1, off"
               :: "v"(lds), "v"((unsigned long long)(uintptr_t)gp) : "memory");
}
__device__ __forceinline__ void wait_async0() {
  asm volatile("s_wait_asynccnt 0x0" ::: "memory");
}
__device__ __forceinline__ void wait_async3() {   // down: 3 asyncs in flight/thread
  asm volatile("s_wait_asynccnt 0x3" ::: "memory");
}
__device__ __forceinline__ void wait_async5() {   // up: 5 asyncs in flight/thread
  asm volatile("s_wait_asynccnt 0x5" ::: "memory");
}
__device__ __forceinline__ unsigned lds_off(const void* p) {
  return (unsigned)(uintptr_t)p;   // AS3 offset survives in low 32 bits
}

// ---------------------------------------------------------------- utilities
__global__ void zero_f32_kernel(float* __restrict__ p, int n) {
  int i = blockIdx.x * blockDim.x + threadIdx.x;
  if (i < n) p[i] = 0.0f;
}

// fp32 -> bf16 flat (for x)
__global__ void cvt_bf16_kernel(const float* __restrict__ src, u16* __restrict__ dst, int n4) {
  int i = blockIdx.x * blockDim.x + threadIdx.x;
  if (i >= n4) return;
  float4 f = reinterpret_cast<const float4*>(src)[i];
  ushort4 o;
  o.x = f2bf(f.x); o.y = f2bf(f.y); o.z = f2bf(f.z); o.w = f2bf(f.w);
  reinterpret_cast<ushort4*>(dst)[i] = o;
}

// fp32 [E][R][C] -> bf16 transposed [E][C][R], 32x32 LDS tiles.
// grid = (C/32, R/32, E), block = 256
__global__ __launch_bounds__(256)
void cvt_transpose_kernel(const float* __restrict__ src, u16* __restrict__ dst,
                          int R, int C) {
  __shared__ u16 tile[32][33];
  int e  = blockIdx.z;
  int c0 = blockIdx.x * 32;
  int r0 = blockIdx.y * 32;
  int tx = threadIdx.x & 7;    // 8 x float4 = 32 cols
  int ty = threadIdx.x >> 3;   // 32 rows
  const float* s = src + ((size_t)e * R + r0 + ty) * C + c0 + tx * 4;
  float4 v = *reinterpret_cast<const float4*>(s);
  tile[tx * 4 + 0][ty] = f2bf(v.x);
  tile[tx * 4 + 1][ty] = f2bf(v.y);
  tile[tx * 4 + 2][ty] = f2bf(v.z);
  tile[tx * 4 + 3][ty] = f2bf(v.w);
  __syncthreads();
  ushort4 o;
  o.x = tile[ty][tx * 4 + 0];
  o.y = tile[ty][tx * 4 + 1];
  o.z = tile[ty][tx * 4 + 2];
  o.w = tile[ty][tx * 4 + 3];
  u16* d = dst + ((size_t)e * C + c0 + ty) * R + r0 + tx * 4;
  *reinterpret_cast<ushort4*>(d) = o;
}

// ---------------------------------------------------------------- routing
__global__ void route_kernel(const int* __restrict__ eidx,
                             int* __restrict__ pairs_sorted,
                             int* __restrict__ off_g,    // E+1
                             int* __restrict__ ntiles_g, // 1
                             int* __restrict__ tile_e,   // MAX_TILES
                             int* __restrict__ tile_s) { // MAX_TILES
  __shared__ int cnt[Ee], off[Ee + 1], cur[Ee];
  int tid = threadIdx.x;
  if (tid < Ee) cnt[tid] = 0;
  __syncthreads();
  for (int p = tid; p < Pp; p += blockDim.x) {
    atomicAdd(&cnt[eidx[p] & (Ee - 1)], 1);
  }
  __syncthreads();
  if (tid == 0) {
    off[0] = 0;
    for (int e = 0; e < Ee; ++e) { off[e + 1] = off[e] + cnt[e]; cur[e] = off[e]; }
    int nt = 0;
    for (int e = 0; e < Ee; ++e)
      for (int r = 0; r < cnt[e]; r += MT) { tile_e[nt] = e; tile_s[nt] = off[e] + r; ++nt; }
    *ntiles_g = nt;
    for (int e = 0; e <= Ee; ++e) off_g[e] = off[e];
  }
  __syncthreads();
  for (int p = tid; p < Pp; p += blockDim.x) {
    int pos = atomicAdd(&cur[eidx[p] & (Ee - 1)], 1);
    pairs_sorted[pos] = p;
  }
}

// ---------------------------------------------------------------- up + GLU
// Block: 32 pairs x 128 H-cols, both GLU halves, double-buffered async pipeline.
// w1t = [E][2H][D] bf16 (K-contiguous rows). grid = (MAX_TILES, H/NB).
__global__ __launch_bounds__(256)
void up_glu_kernel(const u16* __restrict__ xbf,
                   const u16* __restrict__ w1t,
                   u16* __restrict__ actbf,
                   const int* __restrict__ pairs_sorted,
                   const int* __restrict__ off_g,
                   const int* __restrict__ ntiles_g,
                   const int* __restrict__ tile_e,
                   const int* __restrict__ tile_s) {
  int t = blockIdx.x;
  if (t >= *ntiles_g) return;
  int e     = tile_e[t];
  int s0    = tile_s[t];
  int send  = off_g[e + 1];
  int hbase = blockIdx.y * NB;

  __shared__ u16 lA [2][MT][RS];
  __shared__ u16 lBh[2][NB][RS];
  __shared__ u16 lBg[2][NB][RS];

  int tid  = threadIdx.x;
  int lane = tid & 31;
  int wv   = tid >> 5;
  int half = (lane >> 4) & 1;
  int lm   = lane & 15;

  floatx8 accH[2] = {};
  floatx8 accG[2] = {};

  // A staging: uniform, 256 threads x 8B (32 rows x 64B)
  int arow = tid >> 3, aco = (tid & 7) * 4;
  int asp = s0 + arow; if (asp >= send) asp = send - 1;
  const u16* pA = xbf + (size_t)(pairs_sorted[asp] / Kk) * Dd + aco;
  unsigned offA[2] = { lds_off(&lA[0][arow][aco]), lds_off(&lA[1][arow][aco]) };

  // B staging: 2 x b128 per thread per tile (128 rows x 64B)
  int n0 = (tid * 2) >> 2,     c0 = ((tid * 2) & 3) * 8;
  int n1 = (tid * 2 + 1) >> 2, c1 = ((tid * 2 + 1) & 3) * 8;
  const u16* w1h = w1t + ((size_t)e * 2 * Hh + hbase) * Dd;
  const u16* w1g = w1h + (size_t)Hh * Dd;
  const u16* pBh0 = w1h + (size_t)n0 * Dd + c0;
  const u16* pBh1 = w1h + (size_t)n1 * Dd + c1;
  const u16* pBg0 = w1g + (size_t)n0 * Dd + c0;
  const u16* pBg1 = w1g + (size_t)n1 * Dd + c1;
  unsigned offBh0[2] = { lds_off(&lBh[0][n0][c0]), lds_off(&lBh[1][n0][c0]) };
  unsigned offBh1[2] = { lds_off(&lBh[0][n1][c1]), lds_off(&lBh[1][n1][c1]) };
  unsigned offBg0[2] = { lds_off(&lBg[0][n0][c0]), lds_off(&lBg[1][n0][c0]) };
  unsigned offBg1[2] = { lds_off(&lBg[0][n1][c1]), lds_off(&lBg[1][n1][c1]) };

  auto stage = [&](int b) {                 // 5 asyncs/thread, pointers advance KC
    async_b64 (offA[b],   pA);
    async_b128(offBh0[b], pBh0);
    async_b128(offBh1[b], pBh1);
    async_b128(offBg0[b], pBg0);
    async_b128(offBg1[b], pBg1);
    pA += KC; pBh0 += KC; pBh1 += KC; pBg0 += KC; pBg1 += KC;
  };
  int nB = wv * 16 + lm;
  auto compute = [&](int b) {
    Frag a0, a1, bh, bg;
    a0.q[0] = *reinterpret_cast<const uint4*>(&lA[b][lm][half * 8]);
    a0.q[1] = *reinterpret_cast<const uint4*>(&lA[b][lm][16 + half * 8]);
    a1.q[0] = *reinterpret_cast<const uint4*>(&lA[b][16 + lm][half * 8]);
    a1.q[1] = *reinterpret_cast<const uint4*>(&lA[b][16 + lm][16 + half * 8]);
    bh.q[0] = *reinterpret_cast<const uint4*>(&lBh[b][nB][half * 8]);
    bh.q[1] = *reinterpret_cast<const uint4*>(&lBh[b][nB][16 + half * 8]);
    bg.q[0] = *reinterpret_cast<const uint4*>(&lBg[b][nB][half * 8]);
    bg.q[1] = *reinterpret_cast<const uint4*>(&lBg[b][nB][16 + half * 8]);
    accH[0] = __builtin_amdgcn_wmma_f32_16x16x32_bf16(false, a0.v, false, bh.v,
                                                      (short)0, accH[0], false, false);
    accH[1] = __builtin_amdgcn_wmma_f32_16x16x32_bf16(false, a1.v, false, bh.v,
                                                      (short)0, accH[1], false, false);
    accG[0] = __builtin_amdgcn_wmma_f32_16x16x32_bf16(false, a0.v, false, bg.v,
                                                      (short)0, accG[0], false, false);
    accG[1] = __builtin_amdgcn_wmma_f32_16x16x32_bf16(false, a1.v, false, bg.v,
                                                      (short)0, accG[1], false, false);
  };

  const int NSTEP = Dd / KC;                // 32
  stage(0);
  for (int i = 0; i < NSTEP - 1; ++i) {
    stage((i + 1) & 1);                     // overlap next copy with this compute
    wait_async5();                          // in-order completion => batch i landed
    __syncthreads();
    compute(i & 1);
    __syncthreads();                        // protect buffer before batch i+2
  }
  wait_async0();
  __syncthreads();
  compute((NSTEP - 1) & 1);

  // Epilogue: act = gelu_exact(h) * (g + 1), bf16 indexed by sorted slot
  int ncol = hbase + wv * 16 + lm;
  #pragma unroll
  for (int mt = 0; mt < 2; ++mt) {
    #pragma unroll
    for (int r = 0; r < 8; ++r) {
      int m  = mt * 16 + r + half * 8;      // C/D layout: VGPR r -> M=r (+8 hi lanes)
      int sp = s0 + m;
      if (sp < send) {
        float h  = accH[mt][r];
        float g  = accG[mt][r];
        float ge = 0.5f * h * (1.0f + erff(h * 0.70710678118f));
        actbf[(size_t)sp * Hh + ncol] = f2bf(ge * (g + 1.0f));
      }
    }
  }
}

// ---------------------------------------------------------------- down + combine
// Block: 32 pairs x 128 D-cols over K=H, double-buffered. w2t = [E][D][H] bf16.
// grid = (MAX_TILES, D/NB).
__global__ __launch_bounds__(256)
void down_kernel(const u16* __restrict__ actbf,
                 const u16* __restrict__ w2t,
                 const float* __restrict__ gates,
                 float* __restrict__ out,
                 const int* __restrict__ pairs_sorted,
                 const int* __restrict__ off_g,
                 const int* __restrict__ ntiles_g,
                 const int* __restrict__ tile_e,
                 const int* __restrict__ tile_s) {
  int t = blockIdx.x;
  if (t >= *ntiles_g) return;
  int e     = tile_e[t];
  int s0    = tile_s[t];
  int send  = off_g[e + 1];
  int dbase = blockIdx.y * NB;

  __shared__ u16 lA[2][MT][RS];
  __shared__ u16 lB[2][NB][RS];

  int tid  = threadIdx.x;
  int lane = tid & 31;
  int wv   = tid >> 5;
  int half = (lane >> 4) & 1;
  int lm   = lane & 15;

  floatx8 acc[2] = {};

  int arow = tid >> 3, aco = (tid & 7) * 4;
  int asp = s0 + arow; if (asp >= send) asp = send - 1;
  const u16* pA = actbf + (size_t)asp * Hh + aco;
  unsigned offA[2] = { lds_off(&lA[0][arow][aco]), lds_off(&lA[1][arow][aco]) };

  int n0 = (tid * 2) >> 2,     c0 = ((tid * 2) & 3) * 8;
  int n1 = (tid * 2 + 1) >> 2, c1 = ((tid * 2 + 1) & 3) * 8;
  const u16* w2e = w2t + ((size_t)e * Dd + dbase) * Hh;
  const u16* pB0 = w2e + (size_t)n0 * Hh + c0;
  const u16* pB1 = w2e + (size_t)n1 * Hh + c1;
  unsigned offB0[2] = { lds_off(&lB[0][n0][c0]), lds_off(&lB[1][n0][c0]) };
  unsigned offB1[2] = { lds_off(&lB[0][n1][c1]), lds_off(&lB[1][n1][c1]) };

  auto stage = [&](int b) {                 // 3 asyncs/thread
    async_b64 (offA[b],  pA);
    async_b128(offB0[b], pB0);
    async_b128(offB1[b], pB1);
    pA += KC; pB0 += KC; pB1 += KC;
  };
  int nB = wv * 16 + lm;
  auto compute = [&](int b) {
    Frag a0, a1, bb;
    a0.q[0] = *reinterpret_cast<const uint4*>(&lA[b][lm][half * 8]);
    a0.q[1] = *reinterpret_cast<const uint4*>(&lA[b][lm][16 + half * 8]);
    a1.q[0] = *reinterpret_cast<const uint4*>(&lA[b][16 + lm][half * 8]);
    a1.q[1] = *reinterpret_cast<const uint4*>(&lA[b][16 + lm][16 + half * 8]);
    bb.q[0] = *reinterpret_cast<const uint4*>(&lB[b][nB][half * 8]);
    bb.q[1] = *reinterpret_cast<const uint4*>(&lB[b][nB][16 + half * 8]);
    acc[0] = __builtin_amdgcn_wmma_f32_16x16x32_bf16(false, a0.v, false, bb.v,
                                                     (short)0, acc[0], false, false);
    acc[1] = __builtin_amdgcn_wmma_f32_16x16x32_bf16(false, a1.v, false, bb.v,
                                                     (short)0, acc[1], false, false);
  };

  const int NSTEP = Hh / KC;                // 64
  stage(0);
  for (int i = 0; i < NSTEP - 1; ++i) {
    stage((i + 1) & 1);
    wait_async3();
    __syncthreads();
    compute(i & 1);
    __syncthreads();
  }
  wait_async0();
  __syncthreads();
  compute((NSTEP - 1) & 1);

  int ncol = dbase + wv * 16 + lm;
  #pragma unroll
  for (int mt = 0; mt < 2; ++mt) {
    #pragma unroll
    for (int r = 0; r < 8; ++r) {
      int m  = mt * 16 + r + half * 8;
      int sp = s0 + m;
      if (sp < send) {
        int p = pairs_sorted[sp];
        atomicAdd(&out[(size_t)(p / Kk) * Dd + ncol], acc[mt][r] * gates[p]);
      }
    }
  }
}

// ---------------------------------------------------------------- launch
extern "C" void kernel_launch(void* const* d_in, const int* in_sizes, int n_in,
                              void* d_out, int out_size, void* d_ws, size_t ws_size,
                              hipStream_t stream) {
  const float* x    = (const float*)d_in[0];   // [B,S,D]
  const float* ep   = (const float*)d_in[1];   // [B,S,K]
  const int*   eidx = (const int*)  d_in[2];   // [B,S,K]
  const float* W1   = (const float*)d_in[3];   // [E,D,2H]
  const float* W2   = (const float*)d_in[4];   // [E,H,D]
  float*       out  = (float*)d_out;           // [B,S,D]

  // Workspace layout (~143 MB)
  char* ws = (char*)d_ws;
  size_t o = 0;
  u16* xbf  = (u16*)(ws + o); o += (size_t)Tt * Dd * 2;                 // 8 MB
  u16* w1t  = (u16*)(ws + o); o += (size_t)Ee * 2 * Hh * Dd * 2;        // 64 MB [E][2H][D]
  u16* w2t  = (u16*)(ws + o); o += (size_t)Ee * Dd * Hh * 2;            // 32 MB [E][D][H]
  u16* actbf= (u16*)(ws + o); o += (size_t)Pp * Hh * 2;                 // 32 MB
  int* pairs  = (int*)(ws + o); o += (size_t)Pp * 4;
  int* off_g  = (int*)(ws + o); o += (Ee + 1) * 4;
  int* ntiles = (int*)(ws + o); o += 4;
  int* tile_e = (int*)(ws + o); o += MAX_TILES * 4;
  int* tile_s = (int*)(ws + o); o += MAX_TILES * 4;
  (void)ws_size; (void)in_sizes; (void)n_in; (void)out_size;

  // 1) zero output (harness poisons it)
  {
    int n = Tt * Dd;
    zero_f32_kernel<<<(n + 255) / 256, 256, 0, stream>>>(out, n);
  }
  // 2) x fp32 -> bf16 (row-major)
  {
    int n4 = Tt * Dd / 4;
    cvt_bf16_kernel<<<(n4 + 255) / 256, 256, 0, stream>>>(x, xbf, n4);
  }
  // 3) weights fp32 -> bf16 with transpose (K-contiguous B rows)
  {
    dim3 g1((2 * Hh) / 32, Dd / 32, Ee);     // W1 [E][D][2H] -> [E][2H][D]
    cvt_transpose_kernel<<<g1, 256, 0, stream>>>(W1, w1t, Dd, 2 * Hh);
    dim3 g2(Dd / 32, Hh / 32, Ee);           // W2 [E][H][D] -> [E][D][H]
    cvt_transpose_kernel<<<g2, 256, 0, stream>>>(W2, w2t, Hh, Dd);
  }
  // 4) routing
  route_kernel<<<1, 256, 0, stream>>>(eidx, pairs, off_g, ntiles, tile_e, tile_s);
  // 5) grouped up-proj + GLU
  {
    dim3 grid(MAX_TILES, Hh / NB);           // 264 x 16
    up_glu_kernel<<<grid, 256, 0, stream>>>(xbf, w1t, actbf, pairs, off_g,
                                            ntiles, tile_e, tile_s);
  }
  // 6) grouped down-proj + gated combine
  {
    dim3 grid(MAX_TILES, Dd / NB);           // 264 x 8
    down_kernel<<<grid, 256, 0, stream>>>(actbf, w2t, ep, out, pairs, off_g,
                                          ntiles, tile_e, tile_s);
  }
}
